// RNNLanguageModel_67396626808984
// MI455X (gfx1250) — matrix operand order repaired and checked by reading, not compiled, not run on hardware
//
#include <hip/hip_runtime.h>
#include <cstdint>

// ---------------------------------------------------------------------------
// LSTM language model forward for MI455X (gfx1250).
//   tokens[B,T] -> emb gather -> x_proj = X@W + b -> 128 recurrence steps
//   (h@U via WMMA + gate math) -> logits = H@W_cls + b_cls  (268 GFLOP, dominant)
// All GEMMs on v_wmma_f32_16x16x32_bf16; GEMM tiles staged with
// global_load_async_to_lds_b128 + s_wait_asynccnt, double-buffered LDS.
// V=32000 E=256 H=1024 B=32 T=128
// ---------------------------------------------------------------------------

#define VOCAB 32000
#define EDIM  256
#define HDIM  1024
#define BB    32
#define TT    128

typedef __attribute__((ext_vector_type(16))) __bf16 v16bf;
typedef __attribute__((ext_vector_type(8)))  float  v8f;
typedef __attribute__((ext_vector_type(4)))  int    v4i;
typedef unsigned short u16;
typedef unsigned int   u32;

#define AS1 __attribute__((address_space(1)))
#define AS3 __attribute__((address_space(3)))

// Async global->LDS path (ASYNCcnt) if the toolchain exposes the builtins.
// (__has_builtin is 0 in the host pass, which is fine: host never runs these.)
#if __has_builtin(__builtin_amdgcn_global_load_async_to_lds_b128) && \
    __has_builtin(__builtin_amdgcn_s_wait_asynccnt)
#define USE_ASYNC_LDS 1
#else
#define USE_ASYNC_LDS 0
#endif

struct U4x2 { uint4 lo, hi; };

static __device__ __forceinline__ v16bf frag_cast(uint4 lo, uint4 hi) {
  U4x2 t; t.lo = lo; t.hi = hi;
  return __builtin_bit_cast(v16bf, t);
}

static __device__ __forceinline__ u16 f32_to_bf16(float f) {
  u32 u = __builtin_bit_cast(u32, f);
  u32 r = u + 0x7FFFu + ((u >> 16) & 1u);   // round to nearest even
  return (u16)(r >> 16);
}

// Copy 16 bytes global -> LDS (async if available).
static __device__ __forceinline__ void cp16(const u16* g, u16* l) {
#if USE_ASYNC_LDS
  __builtin_amdgcn_global_load_async_to_lds_b128((AS1 v4i*)g, (AS3 v4i*)l, 0, 0);
#else
  *(uint4*)l = *(const uint4*)g;
#endif
}

static __device__ __forceinline__ void wait_async() {
#if USE_ASYNC_LDS
  __builtin_amdgcn_s_wait_asynccnt(0);
#endif
}

// --------------------------- conversion kernels ----------------------------

// s: [rows, cols] f32 row-major  ->  d: [cols, rows] bf16 (transposed).
// LDS-tiled 32x32 so both global streams are coalesced.
__global__ __launch_bounds__(256)
void k_cvt_transpose(const float* __restrict__ s, u16* __restrict__ d,
                     int rows, int cols) {
  __shared__ u16 tile[32][33];
  const int c0 = blockIdx.x * 32;
  const int r0 = blockIdx.y * 32;
  const int tid = threadIdx.x;
#pragma unroll
  for (int i = 0; i < 4; ++i) {
    int idx = tid + i * 256;
    int r = idx >> 5, c = idx & 31;
    tile[r][c] = f32_to_bf16(s[(size_t)(r0 + r) * cols + (c0 + c)]);
  }
  __syncthreads();
#pragma unroll
  for (int i = 0; i < 4; ++i) {
    int idx = tid + i * 256;
    int n = idx >> 5, k = idx & 31;
    d[(size_t)(c0 + n) * rows + (r0 + k)] = tile[k][n];
  }
}

// Xb[(b*T+t)*E + e] = bf16(emb[tokens[b*T+t]*E + e])
__global__ void k_gather_embed(const int* __restrict__ tokens,
                               const float* __restrict__ emb,
                               u16* __restrict__ xb) {
  int i = blockIdx.x * blockDim.x + threadIdx.x;   // B*T*E
  if (i >= BB * TT * EDIM) return;
  int bt = i >> 8;          // E = 256
  int e  = i & 255;
  int tok = tokens[bt];
  xb[i] = f32_to_bf16(emb[(size_t)tok * EDIM + e]);
}

__global__ void k_init_state(u16* __restrict__ h0, float* __restrict__ c0) {
  int i = blockIdx.x * blockDim.x + threadIdx.x;
  if (i < BB * HDIM) { h0[i] = 0; c0[i] = 0.f; }
}

// ------------------------------ WMMA GEMM ----------------------------------
// C[M,N] = A[M,K] @ Bt[N,K]^T + bias[N].  A,Bt bf16 row-major, C f32.
// M,N multiples of 128; K multiple of 32.
// Block = 256 threads (8 waves), tile 128x128; wave tile 64x32 (4x2 WMMAs).
// A and B tiles are both 128x32 row-major -> staged identically with async
// b128 copies into double-buffered LDS (padded stride 40 halves).
__global__ __launch_bounds__(256)
void k_gemm_bf16_wmma(const u16* __restrict__ A, const u16* __restrict__ Bt,
                      const float* __restrict__ bias, float* __restrict__ C,
                      int N, int K) {
  __shared__ __align__(16) u16 As[2][128 * 40];
  __shared__ __align__(16) u16 Bs[2][128 * 40];

  const int tid  = threadIdx.x;
  const int lane = tid & 31;
  const int wid  = tid >> 5;
  const int wm   = wid & 1;          // 0..1 : 64-row half of the tile
  const int wn   = wid >> 1;         // 0..3 : 32-col strip
  const int m0   = blockIdx.y * 128;
  const int n0   = blockIdx.x * 128;
  const int hi   = lane >> 4;        // lane half (ISA fragment layout)
  const int l15  = lane & 15;

  // Per-thread staging coords: 2 x 16B per tile (q = tid, tid+256).
  const int sr0 = tid >> 2,         sk0 = (tid & 3) << 3;
  const int sr1 = (tid + 256) >> 2, sk1 = ((tid + 256) & 3) << 3;

  auto issue_tile = [&](const u16* __restrict__ src, int baserow,
                        u16* __restrict__ dst, int k0) {
    cp16(src + (size_t)(baserow + sr0) * K + (k0 + sk0), dst + sr0 * 40 + sk0);
    cp16(src + (size_t)(baserow + sr1) * K + (k0 + sk1), dst + sr1 * 40 + sk1);
  };

  v8f acc[4][2];
  const v8f vzero = {0.f, 0.f, 0.f, 0.f, 0.f, 0.f, 0.f, 0.f};
#pragma unroll
  for (int a = 0; a < 4; ++a)
#pragma unroll
    for (int b2 = 0; b2 < 2; ++b2) acc[a][b2] = vzero;

  const int nk = K >> 5;
  issue_tile(A,  m0, As[0], 0);
  issue_tile(Bt, n0, Bs[0], 0);

  for (int i = 0; i < nk; ++i) {
    wait_async();            // this wave's staged 16B copies for chunk i done
    __syncthreads();         // chunk i visible to all waves

    const int cb = i & 1;
    if (i + 1 < nk) {        // overlap chunk i+1 transfer with chunk i math
      issue_tile(A,  m0, As[cb ^ 1], (i + 1) << 5);
      issue_tile(Bt, n0, Bs[cb ^ 1], (i + 1) << 5);
    }

    // ---- load all fragments for this chunk, then burst 8 WMMAs ----
    // B fragment: halves 0..7 -> K = h + 8*hi, 8..15 -> K = h+8 + 8*hi, n = l15
    v16bf bfr[2];
#pragma unroll
    for (int b2 = 0; b2 < 2; ++b2) {
      const u16* p = Bs[cb] + (wn * 32 + b2 * 16 + l15) * 40;
      bfr[b2] = frag_cast(*(const uint4*)(p + 8 * hi), *(const uint4*)(p + 16 + 8 * hi));
    }
    // A fragment: halves 0..7 -> K = h + 8*hi, 8..15 -> K = h+8 + 8*hi
    v16bf afr[4];
#pragma unroll
    for (int a = 0; a < 4; ++a) {
      const u16* p = As[cb] + (wm * 64 + a * 16 + l15) * 40;
      afr[a] = frag_cast(*(const uint4*)(p + 8 * hi),
                         *(const uint4*)(p + 16 + 8 * hi));
    }
#pragma unroll
    for (int a = 0; a < 4; ++a)
#pragma unroll
      for (int b2 = 0; b2 < 2; ++b2)
        acc[a][b2] = __builtin_amdgcn_wmma_f32_16x16x32_bf16(
            false, afr[a], false, bfr[b2], (short)0, acc[a][b2], false, false);

    __syncthreads();         // all waves done reading buf before reuse
  }

  // ---- epilogue: C/D layout m = r + 8*hi, n = l15 ----
#pragma unroll
  for (int a = 0; a < 4; ++a) {
#pragma unroll
    for (int b2 = 0; b2 < 2; ++b2) {
      int n = n0 + wn * 32 + b2 * 16 + l15;
      float bv = bias[n];
#pragma unroll
      for (int r = 0; r < 8; ++r) {
        int m = m0 + wm * 64 + a * 16 + (hi ? r + 8 : r);
        C[(size_t)m * N + n] = acc[a][b2][r] + bv;
      }
    }
  }
}

// ------------------------------ LSTM step ----------------------------------
// One timestep: z = x_proj[:,t,:] + h_prev @ U ; gate math ; emit h as bf16.
// Grid = 16 blocks (64 H-columns each); 8 waves = {gate 0..3} x {16-row half}.
// Operands are tiny and L2-hot -> fragments loaded straight from global.
__global__ __launch_bounds__(256)
void k_lstm_step(const u16* __restrict__ hprev, u16* __restrict__ hnext,
                 float* __restrict__ cstate, const u16* __restrict__ Ubt,
                 const float* __restrict__ xproj, u16* __restrict__ hs,
                 int t) {
  __shared__ float zlds[4][32][65];   // [gate][batch row][local col], padded

  const int tid  = threadIdx.x;
  const int lane = tid & 31;
  const int wid  = tid >> 5;
  const int g    = wid & 3;          // gate: i,f,g,o
  const int mt   = wid >> 2;         // 16-row half of the 32 batch rows
  const int j0   = blockIdx.x * 64;
  const int hi   = lane >> 4;
  const int l15  = lane & 15;

  v8f acc[4];
  const v8f vzero = {0.f, 0.f, 0.f, 0.f, 0.f, 0.f, 0.f, 0.f};
#pragma unroll
  for (int s = 0; s < 4; ++s) acc[s] = vzero;

  const int arow = mt * 16 + l15;
  const u16* aptr = hprev + (size_t)arow * HDIM;   // h_prev [32, 1024] bf16

  for (int kc = 0; kc < HDIM; kc += 32) {
    v16bf afr = frag_cast(*(const uint4*)(aptr + kc + 8 * hi),
                          *(const uint4*)(aptr + kc + 16 + 8 * hi));
#pragma unroll
    for (int s = 0; s < 4; ++s) {
      int n = g * HDIM + j0 + s * 16 + l15;        // column of z (gate-major)
      const u16* bptr = Ubt + (size_t)n * HDIM + kc + 8 * hi;  // Ubt [4H, H]
      v16bf bfr = frag_cast(*(const uint4*)bptr, *(const uint4*)(bptr + 16));
      acc[s] = __builtin_amdgcn_wmma_f32_16x16x32_bf16(
          false, afr, false, bfr, (short)0, acc[s], false, false);
    }
  }

  // add x_proj and exchange gate slabs through LDS
#pragma unroll
  for (int s = 0; s < 4; ++s) {
#pragma unroll
    for (int r = 0; r < 8; ++r) {
      int m  = mt * 16 + (hi ? r + 8 : r);         // batch row 0..31
      int jl = s * 16 + l15;                       // local col 0..63
      float v = acc[s][r] +
                xproj[((size_t)m * TT + t) * (4 * HDIM) + g * HDIM + j0 + jl];
      zlds[g][m][jl] = v;
    }
  }
  __syncthreads();

  // elementwise cell update: 2048 elements / 256 threads = 8 each
#pragma unroll
  for (int e = 0; e < 8; ++e) {
    int p  = tid + e * 256;
    int m  = p >> 6;
    int jl = p & 63;
    float zi = zlds[0][m][jl];
    float zf = zlds[1][m][jl];
    float zg = zlds[2][m][jl];
    float zo = zlds[3][m][jl];
    int cj = m * HDIM + j0 + jl;
    float si = 1.f / (1.f + __expf(-zi));
    float sf = 1.f / (1.f + __expf(-zf));
    float so = 1.f / (1.f + __expf(-zo));
    float cn = sf * cstate[cj] + si * tanhf(zg);
    cstate[cj] = cn;
    float h = so * tanhf(cn);
    u16 hb = f32_to_bf16(h);
    hnext[cj] = hb;                                     // next step's A matrix
    hs[((size_t)m * TT + t) * HDIM + j0 + jl] = hb;     // classifier A matrix
  }
}

// ------------------------------- launcher ----------------------------------

extern "C" void kernel_launch(void* const* d_in, const int* in_sizes, int n_in,
                              void* d_out, int out_size, void* d_ws, size_t ws_size,
                              hipStream_t stream) {
  (void)in_sizes; (void)n_in; (void)out_size; (void)ws_size;

  const int*   tokens = (const int*)  d_in[0];
  const float* emb    = (const float*)d_in[1];
  const float* W      = (const float*)d_in[2];
  const float* U      = (const float*)d_in[3];
  const float* bvec   = (const float*)d_in[4];
  const float* Wcls   = (const float*)d_in[5];
  const float* bcls   = (const float*)d_in[6];
  float* out = (float*)d_out;

  char* ws = (char*)d_ws;
  size_t off = 0;
  auto carve = [&](size_t bytes) -> char* {
    char* p = ws + off;
    off = (off + bytes + 255) & ~(size_t)255;
    return p;
  };

  u16*   Wbt   = (u16*)  carve((size_t)4 * HDIM * EDIM * 2);   //  2 MB bf16 W^T  [4096,256]
  u16*   Ubt   = (u16*)  carve((size_t)4 * HDIM * HDIM * 2);   //  8 MB bf16 U^T  [4096,1024]
  u16*   Cbt   = (u16*)  carve((size_t)VOCAB * HDIM * 2);      // 65 MB bf16 Wcls^T [32000,1024]
  u16*   Xb    = (u16*)  carve((size_t)BB * TT * EDIM * 2);    //  2 MB bf16 embeds
  float* xproj = (float*)carve((size_t)BB * TT * 4 * HDIM * 4);// 64 MB f32 x_proj
  u16*   hs    = (u16*)  carve((size_t)BB * TT * HDIM * 2);    //  8 MB bf16 history
  u16*   hA    = (u16*)  carve((size_t)BB * HDIM * 2);
  u16*   hB    = (u16*)  carve((size_t)BB * HDIM * 2);
  float* cs    = (float*)carve((size_t)BB * HDIM * 4);

  // 1) weight conversion to N-major bf16 (LDS-tiled transpose)
  k_cvt_transpose<<<dim3(4 * HDIM / 32, EDIM / 32), 256, 0, stream>>>(
      W, Wbt, EDIM, 4 * HDIM);
  k_cvt_transpose<<<dim3(4 * HDIM / 32, HDIM / 32), 256, 0, stream>>>(
      U, Ubt, HDIM, 4 * HDIM);
  k_cvt_transpose<<<dim3(VOCAB / 32, HDIM / 32), 256, 0, stream>>>(
      Wcls, Cbt, HDIM, VOCAB);
  {
    int n = BB * TT * EDIM;
    k_gather_embed<<<(n + 255) / 256, 256, 0, stream>>>(tokens, emb, Xb);
  }

  // 2) x_proj = X @ W + b   : [4096,256]@[256,4096]
  k_gemm_bf16_wmma<<<dim3(4 * HDIM / 128, BB * TT / 128), 256, 0, stream>>>(
      Xb, Wbt, bvec, xproj, 4 * HDIM, EDIM);

  // 3) recurrence: 128 dependent steps (kernel-launch global sync per step)
  k_init_state<<<(BB * HDIM + 255) / 256, 256, 0, stream>>>(hA, cs);
  u16* hp = hA;
  u16* hn = hB;
  for (int t = 0; t < TT; ++t) {
    k_lstm_step<<<HDIM / 64, 256, 0, stream>>>(hp, hn, cs, Ubt, xproj, hs, t);
    u16* tmp = hp; hp = hn; hn = tmp;
  }

  // 4) logits = hs @ W_cls + b_cls : [4096,1024]@[1024,32000] -> d_out
  k_gemm_bf16_wmma<<<dim3(VOCAB / 128, BB * TT / 128), 256, 0, stream>>>(
      hs, Cbt, bcls, out, BB * TT, VOCAB);
}
